// MLP_30468497998335
// MI455X (gfx1250) — matrix operand order repaired and checked by reading, not compiled, not run on hardware
//
#include <hip/hip_runtime.h>
#include <math.h>

// ---------------------------------------------------------------------------
// MoE-LoRA MLP for MI455X (gfx1250, wave32, WMMA), v3.1: software-pipelined.
//
//   T = 8192 tokens, D = 2048, F = 5632, E = 8, R = 16, ER = 128, TOPK = 2
//
//   * Double-buffered LDS staging of the A operand with prefetch-ahead,
//     using GLOBAL_LOAD_ASYNC_TO_LDS_B64 (ASYNCcnt-tracked) — the builtin
//     takes v2i32 pointers in AS1 (global) / AS3 (LDS), per the compiler
//     diagnostic from the previous round.
//   * B fragments rotated: loads for step i+1 issue before the trailing
//     barrier of step i, overlapping latency with barrier+staging+wmma.
// ---------------------------------------------------------------------------

#define T_TOK 8192
#define DM    2048
#define FF    5632
#define NE    8
#define RK    16
#define ER    128   // NE*RK

typedef __attribute__((ext_vector_type(16))) _Float16 v16h;
typedef __attribute__((ext_vector_type(8)))  _Float16 v8h;
typedef __attribute__((ext_vector_type(8)))  float    v8f;

#define WMMA_F16(a, b, c) \
  __builtin_amdgcn_wmma_f32_16x16x32_f16(false, (a), false, (b), (short)0, (c), false, false)

// ---- async-to-LDS availability (guarded: falls back to reg+ds pipeline) ----
#if defined(__has_builtin)
#  if __has_builtin(__builtin_amdgcn_global_load_async_to_lds_b64) && \
      __has_builtin(__builtin_amdgcn_s_wait_asynccnt)
#    define ASYNC_LDS 1
#  endif
#endif
#ifndef ASYNC_LDS
#  define ASYNC_LDS 0
#endif

// ---- Fragment loaders (CDNA5 ISA 7.12.2 wave32 layouts) --------------------
// A (16x32 f16): lane m<16: row m, K 0-7 & 16-23; lane m>=16: row m-16, K 8-15 & 24-31.
__device__ __forceinline__ v16h load_a_frag(const _Float16* tile, int ld) {
  const int lane = threadIdx.x & 31;
  const int row  = lane & 15;
  const int k    = (lane >> 4) << 3;          // 0 or 8
  const _Float16* p = tile + (size_t)row * ld + k;
  v8h lo = *(const v8h*)(p);
  v8h hi = *(const v8h*)(p + 16);
  v16h r;
#pragma unroll
  for (int i = 0; i < 8; ++i) { r[i] = lo[i]; r[i + 8] = hi[i]; }
  return r;
}

// B (32x16 f16): lane n<16: column n, K 0-7 & 16-23; lane n>=16: column n-16, K 8-15 & 24-31.
__device__ __forceinline__ v16h load_b_frag(const _Float16* wt, int ld) {
  const int lane = threadIdx.x & 31;
  const int n    = lane & 15;
  const int k    = (lane >> 4) << 3;          // 0 or 8
  const _Float16* p = wt + (size_t)n * ld + k;
  v8h lo = *(const v8h*)(p);
  v8h hi = *(const v8h*)(p + 16);
  v16h r;
#pragma unroll
  for (int i = 0; i < 8; ++i) { r[i] = lo[i]; r[i + 8] = hi[i]; }
  return r;
}

// ---- 32x32 f16 tile staging, one b64 slice per thread ----------------------
#if ASYNC_LDS
typedef int v2i_t __attribute__((ext_vector_type(2)));
typedef v2i_t __attribute__((address_space(1)))* gptr_b64;
typedef v2i_t __attribute__((address_space(3)))* lptr_b64;
__device__ __forceinline__ void stage_issue(_Float16* lds, const _Float16* gsrc, int ld) {
  const int q = threadIdx.x, r = q >> 3, c4 = (q & 7) << 2;
  __builtin_amdgcn_global_load_async_to_lds_b64(
      (gptr_b64)(gsrc + (size_t)r * ld + c4),
      (lptr_b64)(lds + r * 32 + c4), 0, 0);
}
#else
__device__ __forceinline__ uint2 fetch_slice(const _Float16* gsrc, int ld) {
  const int q = threadIdx.x, r = q >> 3, c4 = (q & 7) << 2;
  return *(const uint2*)&gsrc[(size_t)r * ld + c4];
}
__device__ __forceinline__ void store_slice(_Float16* lds, uint2 v) {
  const int q = threadIdx.x, r = q >> 3, c4 = (q & 7) << 2;
  *(uint2*)&lds[r * 32 + c4] = v;
}
#endif

// C (16x16 f32): VGPR i -> M = i + 8*(lane>=16), N = lane&15.

// ---------------------------------------------------------------------------
// Stage 0: conversions / packing into f16 workspace
// ---------------------------------------------------------------------------
__global__ __launch_bounds__(256) void k_f32_to_f16(_Float16* __restrict__ dst,
                                                    const float* __restrict__ src,
                                                    size_t n) {
  size_t i = (size_t)blockIdx.x * blockDim.x + threadIdx.x;
  size_t stride = (size_t)gridDim.x * blockDim.x;
  for (; i < n; i += stride) dst[i] = (_Float16)src[i];
}

// src: [E, N, R] f32  ->  dst: [N, E*R] f16  (stacked LoRA-B, K-contiguous)
__global__ __launch_bounds__(256) void k_pack_loraB(_Float16* __restrict__ dst,
                                                    const float* __restrict__ src,
                                                    int N) {
  size_t total = (size_t)N * ER;
  size_t i = (size_t)blockIdx.x * blockDim.x + threadIdx.x;
  size_t stride = (size_t)gridDim.x * blockDim.x;
  for (; i < total; i += stride) {
    int n = (int)(i >> 7);
    int j = (int)(i & 127);
    int e = j >> 4, r = j & 15;
    dst[i] = (_Float16)src[((size_t)e * N + n) * RK + r];
  }
}

// ---------------------------------------------------------------------------
// Stage 1: router.  One wave per token; f32 softmax, top-2, renormalize.
// ---------------------------------------------------------------------------
__global__ __launch_bounds__(256) void k_router(const float* __restrict__ x,
                                                const float* __restrict__ gw,
                                                float* __restrict__ logits_out,
                                                int* __restrict__ sel0,
                                                int* __restrict__ sel1,
                                                float* __restrict__ c0a,
                                                float* __restrict__ c1a) {
  const int wave = threadIdx.x >> 5;
  const int lane = threadIdx.x & 31;
  const int t = blockIdx.x * 8 + wave;
  const float* h = x + (size_t)t * DM;

  float acc[NE];
#pragma unroll
  for (int e = 0; e < NE; ++e) acc[e] = 0.f;
  for (int d = lane; d < DM; d += 32) {
    float hv = h[d];
#pragma unroll
    for (int e = 0; e < NE; ++e) acc[e] += hv * gw[(size_t)e * DM + d];
  }
#pragma unroll
  for (int e = 0; e < NE; ++e)
    for (int off = 16; off > 0; off >>= 1)
      acc[e] += __shfl_xor(acc[e], off, 32);

  if (lane == 0) {
    float m = acc[0];
#pragma unroll
    for (int e = 1; e < NE; ++e) m = fmaxf(m, acc[e]);
    float ex[NE];
#pragma unroll
    for (int e = 0; e < NE; ++e) ex[e] = __expf(acc[e] - m);
    int i0 = 0; float p0 = ex[0];
#pragma unroll
    for (int e = 1; e < NE; ++e) if (ex[e] > p0) { p0 = ex[e]; i0 = e; }
    int i1 = (i0 == 0) ? 1 : 0; float p1 = ex[i1];
#pragma unroll
    for (int e = 0; e < NE; ++e)
      if (e != i0 && ex[e] > p1) { p1 = ex[e]; i1 = e; }
    float inv = 1.f / (p0 + p1);
    sel0[t] = i0; sel1[t] = i1;
    c0a[t] = p0 * inv; c1a[t] = p1 * inv;
#pragma unroll
    for (int e = 0; e < NE; ++e) logits_out[(size_t)t * NE + e] = acc[e];
  }
}

// ---------------------------------------------------------------------------
// Stage 2: LoRA-A projection a = h @ Astackᵀ ([T,128]); masked per-slot f16.
// Block: 32 tokens x 128 cols (wave = expert).  Pipelined A staging.
// ---------------------------------------------------------------------------
__global__ __launch_bounds__(256) void k_loraA(const _Float16* __restrict__ xh,
                                               const _Float16* __restrict__ Ah, // [128, D]
                                               const int* __restrict__ sel0,
                                               const int* __restrict__ sel1,
                                               _Float16* __restrict__ m0,       // [T,128]
                                               _Float16* __restrict__ m1) {
  __shared__ __align__(16) _Float16 abuf[2][32 * 32];
  const int wave = threadIdx.x >> 5;      // expert block 0..7
  const int row0 = blockIdx.x * 32;
  const int col0 = wave * 16;
  v8f acc[2] = {};
  const _Float16* asrc = xh + (size_t)row0 * DM;
  const _Float16* bp = Ah + (size_t)col0 * DM;

#if ASYNC_LDS
  stage_issue(abuf[0], asrc, DM);
#else
  uint2 pre = fetch_slice(asrc, DM);
#endif
  v16h bn = load_b_frag(bp, DM);

  for (int k = 0, it = 0; k < DM; k += 32, ++it) {
    _Float16* cur = abuf[it & 1];
    const bool more = (k + 32 < DM);
#if ASYNC_LDS
    if (more) { stage_issue(abuf[(it + 1) & 1], asrc + k + 32, DM);
                __builtin_amdgcn_s_wait_asynccnt(1); }
    else      { __builtin_amdgcn_s_wait_asynccnt(0); }
#else
    store_slice(cur, pre);
    if (more) pre = fetch_slice(asrc + k + 32, DM);
#endif
    __syncthreads();
    v16h b = bn;
    if (more) bn = load_b_frag(bp + k + 32, DM);   // in flight across barrier
    v16h a0 = load_a_frag(cur, 32);
    v16h a1 = load_a_frag(cur + 16 * 32, 32);
    acc[0] = WMMA_F16(a0, b, acc[0]);
    acc[1] = WMMA_F16(a1, b, acc[1]);
    __syncthreads();
  }
  const int lane = threadIdx.x & 31;
  const int n = lane & 15, mb = (lane >> 4) << 3;
#pragma unroll
  for (int mi = 0; mi < 2; ++mi) {
#pragma unroll
    for (int i = 0; i < 8; ++i) {
      int t = row0 + mi * 16 + mb + i;
      _Float16 v = (_Float16)acc[mi][i];
      _Float16 z = (_Float16)0.f;
      m0[(size_t)t * ER + col0 + n] = (sel0[t] == wave) ? v : z;
      m1[(size_t)t * ER + col0 + n] = (sel1[t] == wave) ? v : z;
    }
  }
}

// ---------------------------------------------------------------------------
// Stage 3: fused W1/W3 GEMMs + LoRA-B corrections + SwiGLU.
// Block: 32 tokens x 256 F; wave: 32x32 macro-tile (2x2 tiles, 8 wmma/step).
// ---------------------------------------------------------------------------
__global__ __launch_bounds__(256) void k_mlp1(const _Float16* __restrict__ xh,
                                              const _Float16* __restrict__ W1h,  // [F,D]
                                              const _Float16* __restrict__ W3h,  // [F,D]
                                              const _Float16* __restrict__ B1h,  // [F,128]
                                              const _Float16* __restrict__ B3h,  // [F,128]
                                              const _Float16* __restrict__ a1m0,
                                              const _Float16* __restrict__ a1m1,
                                              const _Float16* __restrict__ a3m0,
                                              const _Float16* __restrict__ a3m1,
                                              const float* __restrict__ c0a,
                                              const float* __restrict__ c1a,
                                              _Float16* __restrict__ g0,
                                              _Float16* __restrict__ g1,
                                              _Float16* __restrict__ gbar) {
  __shared__ __align__(16) _Float16 abuf[2][32 * 32];
  const int wave = threadIdx.x >> 5;
  const int row0 = blockIdx.x * 32;
  const int f0 = blockIdx.y * 256 + wave * 32;

  v8f ab1[2][2] = {};
  v8f ab3[2][2] = {};
  const _Float16* asrc = xh + (size_t)row0 * DM;
  const _Float16* w1p = W1h + (size_t)f0 * DM;
  const _Float16* w3p = W3h + (size_t)f0 * DM;

#if ASYNC_LDS
  stage_issue(abuf[0], asrc, DM);
#else
  uint2 pre = fetch_slice(asrc, DM);
#endif
  v16h b1n[2], b3n[2];
#pragma unroll
  for (int ni = 0; ni < 2; ++ni) {
    b1n[ni] = load_b_frag(w1p + (size_t)(ni * 16) * DM, DM);
    b3n[ni] = load_b_frag(w3p + (size_t)(ni * 16) * DM, DM);
  }

  for (int k = 0, it = 0; k < DM; k += 32, ++it) {
    _Float16* cur = abuf[it & 1];
    const bool more = (k + 32 < DM);
#if ASYNC_LDS
    if (more) { stage_issue(abuf[(it + 1) & 1], asrc + k + 32, DM);
                __builtin_amdgcn_s_wait_asynccnt(1); }
    else      { __builtin_amdgcn_s_wait_asynccnt(0); }
#else
    store_slice(cur, pre);
    if (more) pre = fetch_slice(asrc + k + 32, DM);
#endif
    __syncthreads();
    v16h b1c[2] = {b1n[0], b1n[1]};
    v16h b3c[2] = {b3n[0], b3n[1]};
    if (more) {
#pragma unroll
      for (int ni = 0; ni < 2; ++ni) {
        b1n[ni] = load_b_frag(w1p + (size_t)(ni * 16) * DM + k + 32, DM);
        b3n[ni] = load_b_frag(w3p + (size_t)(ni * 16) * DM + k + 32, DM);
      }
    }
    v16h a0 = load_a_frag(cur, 32);
    v16h a1 = load_a_frag(cur + 16 * 32, 32);
#pragma unroll
    for (int ni = 0; ni < 2; ++ni) {
      ab1[0][ni] = WMMA_F16(a0, b1c[ni], ab1[0][ni]);
      ab1[1][ni] = WMMA_F16(a1, b1c[ni], ab1[1][ni]);
      ab3[0][ni] = WMMA_F16(a0, b3c[ni], ab3[0][ni]);
      ab3[1][ni] = WMMA_F16(a1, b3c[ni], ab3[1][ni]);
    }
    __syncthreads();
  }

  const int lane = threadIdx.x & 31;
  const int n = lane & 15, mb = (lane >> 4) << 3;

#pragma unroll
  for (int mi = 0; mi < 2; ++mi) {
#pragma unroll
    for (int ni = 0; ni < 2; ++ni) {
      const int rbase = row0 + mi * 16;
      const int fcol = f0 + ni * 16;
      const _Float16* b1p = B1h + (size_t)fcol * ER;
      const _Float16* b3p = B3h + (size_t)fcol * ER;
      const _Float16* a10 = a1m0 + (size_t)rbase * ER;
      const _Float16* a11 = a1m1 + (size_t)rbase * ER;
      const _Float16* a30 = a3m0 + (size_t)rbase * ER;
      const _Float16* a31 = a3m1 + (size_t)rbase * ER;
      v8f c10 = {}, c11 = {}, c30 = {}, c31 = {};
#pragma unroll
      for (int k = 0; k < ER; k += 32) {
        v16h bb1 = load_b_frag(b1p + k, ER);
        v16h bb3 = load_b_frag(b3p + k, ER);
        c10 = WMMA_F16(load_a_frag(a10 + k, ER), bb1, c10);
        c11 = WMMA_F16(load_a_frag(a11 + k, ER), bb1, c11);
        c30 = WMMA_F16(load_a_frag(a30 + k, ER), bb3, c30);
        c31 = WMMA_F16(load_a_frag(a31 + k, ER), bb3, c31);
      }
#pragma unroll
      for (int i = 0; i < 8; ++i) {
        int t = rbase + mb + i;
        float u10 = ab1[mi][ni][i] + c10[i], u30 = ab3[mi][ni][i] + c30[i];
        float u11 = ab1[mi][ni][i] + c11[i], u31 = ab3[mi][ni][i] + c31[i];
        float gg0 = (u10 / (1.f + __expf(-u10))) * u30;   // silu * gate
        float gg1 = (u11 / (1.f + __expf(-u11))) * u31;
        size_t o = (size_t)t * FF + fcol + n;
        g0[o] = (_Float16)gg0;
        g1[o] = (_Float16)gg1;
        gbar[o] = (_Float16)(c0a[t] * gg0 + c1a[t] * gg1);
      }
    }
  }
}

// ---------------------------------------------------------------------------
// Stage 4: t2m = c0·mask_e0(g0@A2allᵀ) + c1·mask_e1(g1@A2allᵀ).
// Block: 32 tokens x 128 cols (wave = expert).  Two pipelined A streams.
// ---------------------------------------------------------------------------
__global__ __launch_bounds__(256) void k_t2(const _Float16* __restrict__ g0,
                                            const _Float16* __restrict__ g1,
                                            const _Float16* __restrict__ A2h,  // [128, F]
                                            const int* __restrict__ sel0,
                                            const int* __restrict__ sel1,
                                            const float* __restrict__ c0a,
                                            const float* __restrict__ c1a,
                                            _Float16* __restrict__ t2m) {      // [T,128]
  __shared__ __align__(16) _Float16 buf0[2][32 * 32];
  __shared__ __align__(16) _Float16 buf1[2][32 * 32];
  const int wave = threadIdx.x >> 5;
  const int row0 = blockIdx.x * 32;
  const int col0 = wave * 16;
  v8f acc0[2] = {}, acc1[2] = {};
  const _Float16* s0 = g0 + (size_t)row0 * FF;
  const _Float16* s1 = g1 + (size_t)row0 * FF;
  const _Float16* bp = A2h + (size_t)col0 * FF;

#if ASYNC_LDS
  stage_issue(buf0[0], s0, FF);
  stage_issue(buf1[0], s1, FF);
#else
  uint2 p0 = fetch_slice(s0, FF);
  uint2 p1 = fetch_slice(s1, FF);
#endif
  v16h bn = load_b_frag(bp, FF);

  for (int k = 0, it = 0; k < FF; k += 32, ++it) {
    _Float16* c0t = buf0[it & 1];
    _Float16* c1t = buf1[it & 1];
    const bool more = (k + 32 < FF);
#if ASYNC_LDS
    if (more) { stage_issue(buf0[(it + 1) & 1], s0 + k + 32, FF);
                stage_issue(buf1[(it + 1) & 1], s1 + k + 32, FF);
                __builtin_amdgcn_s_wait_asynccnt(2); }
    else      { __builtin_amdgcn_s_wait_asynccnt(0); }
#else
    store_slice(c0t, p0);
    store_slice(c1t, p1);
    if (more) { p0 = fetch_slice(s0 + k + 32, FF); p1 = fetch_slice(s1 + k + 32, FF); }
#endif
    __syncthreads();
    v16h b = bn;
    if (more) bn = load_b_frag(bp + k + 32, FF);
    acc0[0] = WMMA_F16(load_a_frag(c0t, 32), b, acc0[0]);
    acc0[1] = WMMA_F16(load_a_frag(c0t + 16 * 32, 32), b, acc0[1]);
    acc1[0] = WMMA_F16(load_a_frag(c1t, 32), b, acc1[0]);
    acc1[1] = WMMA_F16(load_a_frag(c1t + 16 * 32, 32), b, acc1[1]);
    __syncthreads();
  }
  const int lane = threadIdx.x & 31;
  const int n = lane & 15, mb = (lane >> 4) << 3;
#pragma unroll
  for (int mi = 0; mi < 2; ++mi) {
#pragma unroll
    for (int i = 0; i < 8; ++i) {
      int t = row0 + mi * 16 + mb + i;
      float v = 0.f;
      if (sel0[t] == wave) v += c0a[t] * acc0[mi][i];
      if (sel1[t] == wave) v += c1a[t] * acc1[mi][i];
      t2m[(size_t)t * ER + col0 + n] = (_Float16)v;
    }
  }
}

// ---------------------------------------------------------------------------
// Stage 5: out = gbar @ W2ᵀ + t2m @ B2allᵀ  (one accumulator set, two phases)
// Block: 32 tokens x 256 D; wave: 32x32 macro-tile.
// ---------------------------------------------------------------------------
__global__ __launch_bounds__(256) void k_mlp2(const _Float16* __restrict__ gbar,
                                              const _Float16* __restrict__ W2h,  // [D, F]
                                              const _Float16* __restrict__ t2m,  // [T,128]
                                              const _Float16* __restrict__ B2h,  // [D,128]
                                              float* __restrict__ out) {
  __shared__ __align__(16) _Float16 abuf[2][32 * 32];
  const int wave = threadIdx.x >> 5;
  const int row0 = blockIdx.x * 32;
  const int d0 = blockIdx.y * 256 + wave * 32;
  v8f acc[2][2] = {};
  const _Float16* asrc = gbar + (size_t)row0 * FF;
  const _Float16* wp = W2h + (size_t)d0 * FF;

#if ASYNC_LDS
  stage_issue(abuf[0], asrc, FF);
#else
  uint2 pre = fetch_slice(asrc, FF);
#endif
  v16h bnn[2];
#pragma unroll
  for (int ni = 0; ni < 2; ++ni)
    bnn[ni] = load_b_frag(wp + (size_t)(ni * 16) * FF, FF);

  for (int k = 0, it = 0; k < FF; k += 32, ++it) {
    _Float16* cur = abuf[it & 1];
    const bool more = (k + 32 < FF);
#if ASYNC_LDS
    if (more) { stage_issue(abuf[(it + 1) & 1], asrc + k + 32, FF);
                __builtin_amdgcn_s_wait_asynccnt(1); }
    else      { __builtin_amdgcn_s_wait_asynccnt(0); }
#else
    store_slice(cur, pre);
    if (more) pre = fetch_slice(asrc + k + 32, FF);
#endif
    __syncthreads();
    v16h bc[2] = {bnn[0], bnn[1]};
    if (more) {
#pragma unroll
      for (int ni = 0; ni < 2; ++ni)
        bnn[ni] = load_b_frag(wp + (size_t)(ni * 16) * FF + k + 32, FF);
    }
    v16h a0 = load_a_frag(cur, 32);
    v16h a1 = load_a_frag(cur + 16 * 32, 32);
#pragma unroll
    for (int ni = 0; ni < 2; ++ni) {
      acc[0][ni] = WMMA_F16(a0, bc[ni], acc[0][ni]);
      acc[1][ni] = WMMA_F16(a1, bc[ni], acc[1][ni]);
    }
    __syncthreads();
  }
  // LoRA-2 correction: K = 128.
  const _Float16* t2p = t2m + (size_t)row0 * ER;
#pragma unroll
  for (int ni = 0; ni < 2; ++ni) {
    const _Float16* b2p = B2h + (size_t)(d0 + ni * 16) * ER;
#pragma unroll
    for (int k = 0; k < ER; k += 32) {
      v16h bb = load_b_frag(b2p + k, ER);
      acc[0][ni] = WMMA_F16(load_a_frag(t2p + k, ER), bb, acc[0][ni]);
      acc[1][ni] = WMMA_F16(load_a_frag(t2p + (size_t)16 * ER + k, ER), bb, acc[1][ni]);
    }
  }
  const int lane = threadIdx.x & 31;
  const int n = lane & 15, mb = (lane >> 4) << 3;
#pragma unroll
  for (int mi = 0; mi < 2; ++mi) {
#pragma unroll
    for (int ni = 0; ni < 2; ++ni) {
#pragma unroll
      for (int i = 0; i < 8; ++i) {
        out[(size_t)(row0 + mi * 16 + mb + i) * DM + d0 + ni * 16 + n] = acc[mi][ni][i];
      }
    }
  }
}

// ---------------------------------------------------------------------------
extern "C" void kernel_launch(void* const* d_in, const int* in_sizes, int n_in,
                              void* d_out, int out_size, void* d_ws, size_t ws_size,
                              hipStream_t stream) {
  (void)in_sizes; (void)n_in; (void)out_size; (void)ws_size;
  const float* x    = (const float*)d_in[0];
  const float* gate = (const float*)d_in[1];
  const float* W1   = (const float*)d_in[2];
  const float* W3   = (const float*)d_in[3];
  const float* W2   = (const float*)d_in[4];
  const float* A1   = (const float*)d_in[5];
  const float* B1   = (const float*)d_in[6];
  const float* A3   = (const float*)d_in[7];
  const float* B3   = (const float*)d_in[8];
  const float* A2   = (const float*)d_in[9];
  const float* B2   = (const float*)d_in[10];

  float* out = (float*)d_out;
  float* logits = out + (size_t)T_TOK * DM;

  // ---- workspace carve-up (~400 MB) ----
  char* p = (char*)d_ws;
  auto carve = [&](size_t bytes) {
    char* r = p;
    p += (bytes + 255) & ~(size_t)255;
    return r;
  };
  _Float16* xh   = (_Float16*)carve((size_t)T_TOK * DM * 2);
  _Float16* W1h  = (_Float16*)carve((size_t)FF * DM * 2);
  _Float16* W3h  = (_Float16*)carve((size_t)FF * DM * 2);
  _Float16* W2h  = (_Float16*)carve((size_t)DM * FF * 2);
  _Float16* A1h  = (_Float16*)carve((size_t)ER * DM * 2);
  _Float16* A3h  = (_Float16*)carve((size_t)ER * DM * 2);
  _Float16* A2h  = (_Float16*)carve((size_t)ER * FF * 2);
  _Float16* B1h  = (_Float16*)carve((size_t)FF * ER * 2);
  _Float16* B3h  = (_Float16*)carve((size_t)FF * ER * 2);
  _Float16* B2h  = (_Float16*)carve((size_t)DM * ER * 2);
  _Float16* a1m0 = (_Float16*)carve((size_t)T_TOK * ER * 2);
  _Float16* a1m1 = (_Float16*)carve((size_t)T_TOK * ER * 2);
  _Float16* a3m0 = (_Float16*)carve((size_t)T_TOK * ER * 2);
  _Float16* a3m1 = (_Float16*)carve((size_t)T_TOK * ER * 2);
  _Float16* g0   = (_Float16*)carve((size_t)T_TOK * FF * 2);
  _Float16* g1   = (_Float16*)carve((size_t)T_TOK * FF * 2);
  _Float16* gbar = (_Float16*)carve((size_t)T_TOK * FF * 2);
  _Float16* t2m  = (_Float16*)carve((size_t)T_TOK * ER * 2);
  int*   sel0 = (int*)carve((size_t)T_TOK * 4);
  int*   sel1 = (int*)carve((size_t)T_TOK * 4);
  float* c0a  = (float*)carve((size_t)T_TOK * 4);
  float* c1a  = (float*)carve((size_t)T_TOK * 4);

  auto gsz = [](size_t n) {
    size_t g = (n + 255) / 256;
    return (unsigned)(g > 4096 ? 4096 : g);
  };

  // Stage 0: f16 staging
  k_f32_to_f16<<<gsz((size_t)T_TOK * DM), 256, 0, stream>>>(xh, x, (size_t)T_TOK * DM);
  k_f32_to_f16<<<gsz((size_t)FF * DM), 256, 0, stream>>>(W1h, W1, (size_t)FF * DM);
  k_f32_to_f16<<<gsz((size_t)FF * DM), 256, 0, stream>>>(W3h, W3, (size_t)FF * DM);
  k_f32_to_f16<<<gsz((size_t)DM * FF), 256, 0, stream>>>(W2h, W2, (size_t)DM * FF);
  k_f32_to_f16<<<gsz((size_t)ER * DM), 256, 0, stream>>>(A1h, A1, (size_t)ER * DM);
  k_f32_to_f16<<<gsz((size_t)ER * DM), 256, 0, stream>>>(A3h, A3, (size_t)ER * DM);
  k_f32_to_f16<<<gsz((size_t)ER * FF), 256, 0, stream>>>(A2h, A2, (size_t)ER * FF);
  k_pack_loraB<<<gsz((size_t)FF * ER), 256, 0, stream>>>(B1h, B1, FF);
  k_pack_loraB<<<gsz((size_t)FF * ER), 256, 0, stream>>>(B3h, B3, FF);
  k_pack_loraB<<<gsz((size_t)DM * ER), 256, 0, stream>>>(B2h, B2, DM);

  // Stage 1: routing
  k_router<<<T_TOK / 8, 256, 0, stream>>>(x, gate, logits, sel0, sel1, c0a, c1a);

  // Stage 2: LoRA-A projections (masked per top-k slot)
  k_loraA<<<T_TOK / 32, 256, 0, stream>>>(xh, A1h, sel0, sel1, a1m0, a1m1);
  k_loraA<<<T_TOK / 32, 256, 0, stream>>>(xh, A3h, sel0, sel1, a3m0, a3m1);

  // Stage 3: fused W1/W3 + LoRA-B + SwiGLU
  k_mlp1<<<dim3(T_TOK / 32, FF / 256), 256, 0, stream>>>(
      xh, W1h, W3h, B1h, B3h, a1m0, a1m1, a3m0, a3m1, c0a, c1a, g0, g1, gbar);

  // Stage 4: A2 projection, masked + coef-weighted
  k_t2<<<T_TOK / 32, 256, 0, stream>>>(g0, g1, A2h, sel0, sel1, c0a, c1a, t2m);

  // Stage 5: W2 + LoRA-B2 fused
  k_mlp2<<<dim3(T_TOK / 32, DM / 256), 256, 0, stream>>>(gbar, W2h, t2m, B2h, out);
}
